// MLA_8074538516565
// MI455X (gfx1250) — compile-verified, matrix operand order checked by reading, not run on hardware
//
#include <hip/hip_runtime.h>
#include <math.h>

// ---------------------------------------------------------------------------
// MLA forward for MI455X (gfx1250, wave32, WMMA).
// The run is HBM-bound on materializing w (B,H,S,S) ~1.07GB fp32, so the
// design goals are: (1) f16 WMMA with f32 accumulation for all GEMMs
// (inputs are O(1) magnitude -> f16 safe, more mantissa than bf16),
// (2) minimum passes over w: score GEMM skips tiles fully above the causal
// diagonal, softmax reads each element once (row cached in LDS), and w@v
// skips K-tiles beyond the diagonal where w == 0 exactly,
// (3) software-pipelined, double-buffered LDS staging in the GEMM so global
// load latency overlaps the ds_load/wmma work (one barrier per K-step).
// ---------------------------------------------------------------------------

#define DMODEL 2048
#define NHEADS 16
#define DCQ    1536
#define DC     512
#define DH     128
#define DR     64
#define BBATCH 4
#define SLEN   2048

typedef __attribute__((ext_vector_type(16))) _Float16 v16h;
typedef __attribute__((ext_vector_type(8)))  _Float16 v8h;
typedef __attribute__((ext_vector_type(8)))  float    v8f;

union HFrag { v16h v; v8h h[2]; };

// ---------------------------------------------------------------------------
// Generic batched WMMA GEMM:  C = alpha * A(MxK) @ B(KxN) + bias
//   BT    : B operand is given as [N][K] row-major (i.e. we need B^T)
//   CSKIP : skip output tiles fully above the causal diagonal (score GEMM)
//   CK    : A is lower-triangular (w): skip K-tiles with k > m0+63
// C batch offset = (z/cdiv)*chi + (z%cdiv)*clo  (supports [B,S,H*DH] scatter)
// All dims are multiples of the tile sizes in this problem (no edge guards).
// ---------------------------------------------------------------------------
template<bool BT, bool CSKIP, bool CK>
__global__ __launch_bounds__(256)
void mla_gemm_wmma(const float* __restrict__ A, int lda, long long batchA,
                   const float* __restrict__ B, int ldb, long long batchB,
                   const float* __restrict__ bias,
                   float* __restrict__ C, int ldc,
                   int cdiv, long long chi, long long clo,
                   int M, int N, int K, float alpha)
{
  // double-buffered tiles; padded rows (80B stride) break ds bank conflicts
  __shared__ _Float16 As[2][64][40];
  __shared__ _Float16 Bs[2][64][40];   // stored as Bs[.][n][k]

  const int z = blockIdx.z;
  A += (size_t)z * batchA;
  B += (size_t)z * batchB;
  C += (size_t)(z / cdiv) * chi + (size_t)(z % cdiv) * clo;

  const int m0 = blockIdx.y * 64;
  const int n0 = blockIdx.x * 64;
  if (CSKIP && n0 > m0 + 63) return;   // fully masked tile: softmax writes 0s

  const int t    = threadIdx.x;
  const int lane = t & 31;
  const int wave = t >> 5;
  const int wm   = wave >> 1;          // 0..3 : 16-row slice
  const int wn   = wave & 1;           // 0..1 : 32-col slice
  const int g    = lane >> 4;          // K-half select per WMMA layout
  const int mr   = lane & 15;

  // staging indices (each thread moves 2 float4 per operand per K-step)
  const int ra = t >> 3;               // 0..31 row (A / B-transposed)
  const int qa = (t & 7) * 4;          // K quad
  const int kb = t >> 4;               // 0..15 k (B row-major)
  const int qb = (t & 15) * 4;         // N quad

  v8f acc0 = {};
  v8f acc1 = {};

  const int kEnd = CK ? ((K < m0 + 64) ? K : (m0 + 64)) : K;

  float4 fa0, fa1, fb0, fb1;

  auto loadTile = [&](int k0) {
    fa0 = *reinterpret_cast<const float4*>(&A[(size_t)(m0 + ra)      * lda + (k0 + qa)]);
    fa1 = *reinterpret_cast<const float4*>(&A[(size_t)(m0 + ra + 32) * lda + (k0 + qa)]);
    if (BT) {
      fb0 = *reinterpret_cast<const float4*>(&B[(size_t)(n0 + ra)      * ldb + (k0 + qa)]);
      fb1 = *reinterpret_cast<const float4*>(&B[(size_t)(n0 + ra + 32) * ldb + (k0 + qa)]);
    } else {
      fb0 = *reinterpret_cast<const float4*>(&B[(size_t)(k0 + kb)      * ldb + (n0 + qb)]);
      fb1 = *reinterpret_cast<const float4*>(&B[(size_t)(k0 + kb + 16) * ldb + (n0 + qb)]);
    }
  };

  auto storeTile = [&](int buf) {
    _Float16* d0 = &As[buf][ra][qa];
    d0[0] = (_Float16)fa0.x; d0[1] = (_Float16)fa0.y;
    d0[2] = (_Float16)fa0.z; d0[3] = (_Float16)fa0.w;
    _Float16* d1 = &As[buf][ra + 32][qa];
    d1[0] = (_Float16)fa1.x; d1[1] = (_Float16)fa1.y;
    d1[2] = (_Float16)fa1.z; d1[3] = (_Float16)fa1.w;
    if (BT) {
      _Float16* e0 = &Bs[buf][ra][qa];
      e0[0] = (_Float16)fb0.x; e0[1] = (_Float16)fb0.y;
      e0[2] = (_Float16)fb0.z; e0[3] = (_Float16)fb0.w;
      _Float16* e1 = &Bs[buf][ra + 32][qa];
      e1[0] = (_Float16)fb1.x; e1[1] = (_Float16)fb1.y;
      e1[2] = (_Float16)fb1.z; e1[3] = (_Float16)fb1.w;
    } else {           // transpose while staging: Bs[n][k]
      Bs[buf][qb + 0][kb] = (_Float16)fb0.x;
      Bs[buf][qb + 1][kb] = (_Float16)fb0.y;
      Bs[buf][qb + 2][kb] = (_Float16)fb0.z;
      Bs[buf][qb + 3][kb] = (_Float16)fb0.w;
      Bs[buf][qb + 0][kb + 16] = (_Float16)fb1.x;
      Bs[buf][qb + 1][kb + 16] = (_Float16)fb1.y;
      Bs[buf][qb + 2][kb + 16] = (_Float16)fb1.z;
      Bs[buf][qb + 3][kb + 16] = (_Float16)fb1.w;
    }
  };

  loadTile(0);
  int buf = 0;
  for (int k0 = 0; k0 < kEnd; k0 += 32) {
    storeTile(buf);
    __syncthreads();                     // one barrier per K-step (dbl buffer)
    if (k0 + 32 < kEnd) loadTile(k0 + 32);   // next tile in flight during wmma

    HFrag a, b0, b1;
    a.h[0]  = *(const v8h*)&As[buf][wm * 16 + mr][g * 8];
    a.h[1]  = *(const v8h*)&As[buf][wm * 16 + mr][16 + g * 8];
    b0.h[0] = *(const v8h*)&Bs[buf][wn * 32 + mr][g * 8];
    b0.h[1] = *(const v8h*)&Bs[buf][wn * 32 + mr][16 + g * 8];
    b1.h[0] = *(const v8h*)&Bs[buf][wn * 32 + 16 + mr][g * 8];
    b1.h[1] = *(const v8h*)&Bs[buf][wn * 32 + 16 + mr][16 + g * 8];

    acc0 = __builtin_amdgcn_wmma_f32_16x16x32_f16(false, a.v, false, b0.v,
                                                  (short)0, acc0, false, false);
    acc1 = __builtin_amdgcn_wmma_f32_16x16x32_f16(false, a.v, false, b1.v,
                                                  (short)0, acc1, false, false);
    buf ^= 1;
  }

  // ---- epilogue: C/D layout VGPR r -> row r + 8*(lane/16), col = lane%16 ----
  const int rowBase = m0 + wm * 16 + g * 8;
  const int col0 = n0 + wn * 32 + mr;
  const int col1 = col0 + 16;
  const float bv0 = bias ? bias[col0] : 0.0f;
  const float bv1 = bias ? bias[col1] : 0.0f;
  #pragma unroll
  for (int r = 0; r < 8; ++r) {
    C[(size_t)(rowBase + r) * ldc + col0] = alpha * acc0[r] + bv0;
    C[(size_t)(rowBase + r) * ldc + col1] = alpha * acc1[r] + bv1;
  }
}

// ---------------------------------------------------------------------------
// In-place RMSNorm over rows: x *= rsqrt(mean(x^2)+eps) * g
// ---------------------------------------------------------------------------
__global__ __launch_bounds__(256)
void mla_rmsnorm(float* __restrict__ x, const float* __restrict__ g, int cols)
{
  float* p = x + (size_t)blockIdx.x * cols;
  const int tid = threadIdx.x;
  __shared__ float red[256];
  float s = 0.0f;
  for (int c = tid; c < cols; c += 256) { const float v = p[c]; s += v * v; }
  red[tid] = s; __syncthreads();
  for (int off = 128; off > 0; off >>= 1) {
    if (tid < off) red[tid] += red[tid + off];
    __syncthreads();
  }
  const float inv = rsqrtf(red[0] / (float)cols + 1e-6f);
  for (int c = tid; c < cols; c += 256) p[c] = p[c] * inv * g[c];
}

// ---------------------------------------------------------------------------
// kR: in-place RMSNorm(64) + RoPE. One wave per (b,s) row; lane i owns pair i.
// ---------------------------------------------------------------------------
__global__ __launch_bounds__(32)
void mla_finalize_kr(float* __restrict__ kr, const float* __restrict__ g,
                     const float* __restrict__ freqs, int S)
{
  const int row  = blockIdx.x;          // b*S + s
  const int s    = row % S;
  const int lane = threadIdx.x;         // 0..31
  float* p = kr + (size_t)row * 64;
  float a = p[2 * lane], b = p[2 * lane + 1];
  float ss = a * a + b * b;
  for (int off = 16; off > 0; off >>= 1) ss += __shfl_xor(ss, off, 32);
  const float inv = rsqrtf(ss / 64.0f + 1e-6f);
  a = a * inv * g[2 * lane];
  b = b * inv * g[2 * lane + 1];
  const float cs = freqs[(s * 32 + lane) * 2 + 0];
  const float sn = freqs[(s * 32 + lane) * 2 + 1];
  p[2 * lane]     = a * cs - b * sn;
  p[2 * lane + 1] = a * sn + b * cs;
}

// ---------------------------------------------------------------------------
// q[B,H,S,192] = concat(qC, rope(qR))   (reference order: qC first!)
// ---------------------------------------------------------------------------
__global__ __launch_bounds__(192)
void mla_assemble_q(const float* __restrict__ qC, const float* __restrict__ qR,
                    const float* __restrict__ freqs, float* __restrict__ q, int S)
{
  const int idx = blockIdx.x;
  const int h  = idx % NHEADS;
  const int bs = idx / NHEADS;
  const int s  = bs % S;
  const int b  = bs / S;
  const int d  = threadIdx.x;
  float* out = q + (((size_t)(b * NHEADS + h)) * S + s) * 192;
  if (d < DH) {
    out[d] = qC[(size_t)bs * (NHEADS * DH) + h * DH + d];
  } else {
    const int j = d - DH;
    const int i = j >> 1;
    const float a  = qR[(size_t)bs * (NHEADS * DR) + h * DR + 2 * i];
    const float bb = qR[(size_t)bs * (NHEADS * DR) + h * DR + 2 * i + 1];
    const float cs = freqs[(s * 32 + i) * 2 + 0];
    const float sn = freqs[(s * 32 + i) * 2 + 1];
    out[d] = (j & 1) ? (a * sn + bb * cs) : (a * cs - bb * sn);
  }
}

// ---------------------------------------------------------------------------
// k[B,H,S,192] = concat(kR_rope (broadcast over h), kC)   (kR first!)
// ---------------------------------------------------------------------------
__global__ __launch_bounds__(192)
void mla_assemble_k(const float* __restrict__ kr, const float* __restrict__ kC,
                    float* __restrict__ k, int S)
{
  const int idx = blockIdx.x;
  const int h  = idx % NHEADS;
  const int bs = idx / NHEADS;
  const int s  = bs % S;
  const int b  = bs / S;
  const int d  = threadIdx.x;
  float* out = k + (((size_t)(b * NHEADS + h)) * S + s) * 192;
  if (d < DR) out[d] = kr[(size_t)bs * DR + d];
  else        out[d] = kC[(size_t)bs * (NHEADS * DH) + h * DH + (d - DR)];
}

// ---------------------------------------------------------------------------
// v[B,H,S,128] <- vC[(b,s), h*128+d]   (pure layout transpose)
// ---------------------------------------------------------------------------
__global__ __launch_bounds__(256)
void mla_assemble_v(const float* __restrict__ vC, float* __restrict__ v,
                    int S, int total)
{
  const int i = blockIdx.x * 256 + threadIdx.x;
  if (i >= total) return;
  const int d  = i % DH;
  const int t  = i / DH;
  const int s  = t % S;
  const int bh = t / S;
  const int h  = bh % NHEADS;
  const int b  = bh / NHEADS;
  v[i] = vC[((size_t)(b * S + s)) * (NHEADS * DH) + h * DH + d];
}

// ---------------------------------------------------------------------------
// Causal softmax over rows of w[z, r, :]. Single pass over HBM: valid row
// elements are cached in LDS (8KB), exp computed once, masked region gets
// exact 0 (matches exp(finfo.min - max) == 0 in the reference).
// ---------------------------------------------------------------------------
__global__ __launch_bounds__(256)
void mla_softmax_causal(float* __restrict__ w, int S)
{
  __shared__ float red[256];
  __shared__ float ex[SLEN];

  const int r = blockIdx.x;
  float* p = w + ((size_t)blockIdx.y * S + r) * S;
  const int n = r + 1;
  const int tid = threadIdx.x;

  float mx = -3.402823466e38f;
  for (int c = tid; c < n; c += 256) {
    const float vv = p[c];               // only HBM read of this element
    ex[c] = vv;
    mx = fmaxf(mx, vv);
  }
  red[tid] = mx; __syncthreads();
  for (int off = 128; off > 0; off >>= 1) {
    if (tid < off) red[tid] = fmaxf(red[tid], red[tid + off]);
    __syncthreads();
  }
  mx = red[0];
  __syncthreads();

  float sm = 0.0f;
  for (int c = tid; c < n; c += 256) {   // each thread owns its ex[] slots
    const float e = __expf(ex[c] - mx);
    ex[c] = e;
    sm += e;
  }
  red[tid] = sm; __syncthreads();
  for (int off = 128; off > 0; off >>= 1) {
    if (tid < off) red[tid] += red[tid + off];
    __syncthreads();
  }
  const float inv = 1.0f / red[0];

  for (int c = tid; c < S; c += 256)
    p[c] = (c < n) ? ex[c] * inv : 0.0f;
}

// ---------------------------------------------------------------------------
extern "C" void kernel_launch(void* const* d_in, const int* in_sizes, int n_in,
                              void* d_out, int out_size, void* d_ws, size_t ws_size,
                              hipStream_t stream)
{
  (void)in_sizes; (void)n_in; (void)out_size; (void)ws_size;

  const float* h        = (const float*)d_in[0];
  const float* freqs    = (const float*)d_in[1];
  // d_in[2] = mask (causal, known analytically) — unused
  const float* Wq_down  = (const float*)d_in[3];
  const float* bq_down  = (const float*)d_in[4];
  const float* gq_norm  = (const float*)d_in[5];
  const float* Wqc      = (const float*)d_in[6];
  const float* bqc      = (const float*)d_in[7];
  const float* Wqr      = (const float*)d_in[8];
  const float* bqr      = (const float*)d_in[9];
  const float* Wkr      = (const float*)d_in[10];
  const float* bkr      = (const float*)d_in[11];
  const float* gkr_norm = (const float*)d_in[12];
  const float* Wkv_down = (const float*)d_in[13];
  const float* bkv_down = (const float*)d_in[14];
  const float* gkv_norm = (const float*)d_in[15];
  const float* Wkc      = (const float*)d_in[16];
  const float* bkc      = (const float*)d_in[17];
  const float* Wvc      = (const float*)d_in[18];
  const float* bvc      = (const float*)d_in[19];
  const float* Wo       = (const float*)d_in[20];
  const float* bo       = (const float*)d_in[21];

  const int M = BBATCH * SLEN;                  // 8192 token rows
  float* ws = (float*)d_ws;
  size_t o = 0;
  float* cQ  = ws + o; o += (size_t)M * DCQ;                     // 12.6M
  float* cKV = ws + o; o += (size_t)M * DC;                      //  4.2M
  float* kR  = ws + o; o += (size_t)M * DR;                      //  0.5M
  float* qC  = ws + o; o += (size_t)M * (NHEADS * DH);           // 16.8M
  float* qR  = ws + o; o += (size_t)M * (NHEADS * DR);           //  8.4M
  float* kC  = ws + o; o += (size_t)M * (NHEADS * DH);           // 16.8M
  float* vC  = ws + o; o += (size_t)M * (NHEADS * DH);           // 16.8M
  float* q   = ws + o; o += (size_t)BBATCH * NHEADS * SLEN * 192;// 25.2M
  float* k   = ws + o; o += (size_t)BBATCH * NHEADS * SLEN * 192;// 25.2M
  float* v   = ws + o; o += (size_t)BBATCH * NHEADS * SLEN * DH; // 16.8M
  float* attn = qC;  // alias: qC is dead after assemble_q

  const dim3 blk(256);

  // ---- down projections + norms ----
  mla_gemm_wmma<false,false,false><<<dim3(DCQ/64, M/64, 1), blk, 0, stream>>>(
      h, DMODEL, 0LL, Wq_down, DCQ, 0LL, bq_down,
      cQ, DCQ, 1, 0LL, 0LL, M, DCQ, DMODEL, 1.0f);
  mla_rmsnorm<<<M, 256, 0, stream>>>(cQ, gq_norm, DCQ);

  mla_gemm_wmma<false,false,false><<<dim3(DC/64, M/64, 1), blk, 0, stream>>>(
      h, DMODEL, 0LL, Wkv_down, DC, 0LL, bkv_down,
      cKV, DC, 1, 0LL, 0LL, M, DC, DMODEL, 1.0f);
  mla_rmsnorm<<<M, 256, 0, stream>>>(cKV, gkv_norm, DC);

  mla_gemm_wmma<false,false,false><<<dim3(DR/64, M/64, 1), blk, 0, stream>>>(
      h, DMODEL, 0LL, Wkr, DR, 0LL, bkr,
      kR, DR, 1, 0LL, 0LL, M, DR, DMODEL, 1.0f);
  mla_finalize_kr<<<M, 32, 0, stream>>>(kR, gkr_norm, freqs, SLEN);

  // ---- up projections ----
  mla_gemm_wmma<false,false,false><<<dim3((NHEADS*DH)/64, M/64, 1), blk, 0, stream>>>(
      cQ, DCQ, 0LL, Wqc, NHEADS*DH, 0LL, bqc,
      qC, NHEADS*DH, 1, 0LL, 0LL, M, NHEADS*DH, DCQ, 1.0f);
  mla_gemm_wmma<false,false,false><<<dim3((NHEADS*DR)/64, M/64, 1), blk, 0, stream>>>(
      cQ, DCQ, 0LL, Wqr, NHEADS*DR, 0LL, bqr,
      qR, NHEADS*DR, 1, 0LL, 0LL, M, NHEADS*DR, DCQ, 1.0f);
  mla_gemm_wmma<false,false,false><<<dim3((NHEADS*DH)/64, M/64, 1), blk, 0, stream>>>(
      cKV, DC, 0LL, Wkc, NHEADS*DH, 0LL, bkc,
      kC, NHEADS*DH, 1, 0LL, 0LL, M, NHEADS*DH, DC, 1.0f);
  mla_gemm_wmma<false,false,false><<<dim3((NHEADS*DH)/64, M/64, 1), blk, 0, stream>>>(
      cKV, DC, 0LL, Wvc, NHEADS*DH, 0LL, bvc,
      vC, NHEADS*DH, 1, 0LL, 0LL, M, NHEADS*DH, DC, 1.0f);

  // ---- assemble q/k/v in [B,H,S,d] layouts ----
  mla_assemble_q<<<M * NHEADS, 192, 0, stream>>>(qC, qR, freqs, q, SLEN);
  mla_assemble_k<<<M * NHEADS, 192, 0, stream>>>(kR, kC, k, SLEN);
  {
    const int total = BBATCH * NHEADS * SLEN * DH;
    mla_assemble_v<<<(total + 255) / 256, 256, 0, stream>>>(vC, v, SLEN, total);
  }

  // ---- attention: scores -> softmax -> w@v ----
  float* wout = (float*)d_out + (size_t)M * DMODEL;   // w right after h_out
  const float scale = 1.0f / sqrtf((float)(DH + DR));
  const int ZB = BBATCH * NHEADS;

  // w = scale * q @ k^T  (BT=true, skip fully-masked tiles)
  mla_gemm_wmma<true,true,false><<<dim3(SLEN/64, SLEN/64, ZB), blk, 0, stream>>>(
      q, 192, (long long)SLEN * 192, k, 192, (long long)SLEN * 192, nullptr,
      wout, SLEN, 1, (long long)SLEN * SLEN, 0LL, SLEN, SLEN, 192, scale);

  mla_softmax_causal<<<dim3(SLEN, ZB), blk, 0, stream>>>(wout, SLEN);

  // attn[B,S,H*128] = w @ v   (CK=true: w is lower-triangular)
  mla_gemm_wmma<false,false,true><<<dim3(DH/64, SLEN/64, ZB), blk, 0, stream>>>(
      wout, SLEN, (long long)SLEN * SLEN, v, DH, (long long)SLEN * DH, nullptr,
      attn, DMODEL, NHEADS, (long long)SLEN * DMODEL, (long long)DH,
      SLEN, DH, SLEN, 1.0f);

  // ---- h_out = attn @ Wo + bo ----
  mla_gemm_wmma<false,false,false><<<dim3(DMODEL/64, M/64, 1), blk, 0, stream>>>(
      attn, DMODEL, 0LL, Wo, DMODEL, 0LL, bo,
      (float*)d_out, DMODEL, 1, 0LL, 0LL, M, DMODEL, DMODEL, 1.0f);
}
